// ProDAR_NN_86543591014542
// MI455X (gfx1250) — compile-verified, matrix operand order checked by reading, not run on hardware
//
#include <hip/hip_runtime.h>
#include <hip/hip_bf16.h>

typedef __attribute__((ext_vector_type(16))) _Float16 v16h;
typedef __attribute__((ext_vector_type(8)))  float    v8f;

// ---------------------------------------------------------------------------
// Degree: deg[col[e]] += 1 for every edge
// ---------------------------------------------------------------------------
__global__ void deg_kernel(const int* __restrict__ col, float* __restrict__ degacc, int E_) {
    int e = blockIdx.x * blockDim.x + threadIdx.x;
    if (e < E_) atomicAdd(&degacc[col[e]], 1.0f);
}

// dis = rsqrt(deg+1), snorm = 1/(deg+1)
__global__ void norm_kernel(const float* __restrict__ degacc,
                            float* __restrict__ dis, float* __restrict__ snorm, int n) {
    int i = blockIdx.x * blockDim.x + threadIdx.x;
    if (i < n) {
        float d = degacc[i] + 1.0f;
        dis[i]   = rsqrtf(d);
        snorm[i] = 1.0f / d;
    }
}

// ---------------------------------------------------------------------------
// Pack a row-major f32 weight matrix B[K, Ncols] into f16 in the exact WMMA
// B-operand per-lane element order, zero-filling k >= K and n >= Ncols.
// Layout: out[((nt*ksteps + ks)*32 + lane)*16 + i],
//   k = ks*32 + ((i&8)?16:0) + (lane>>4)*8 + (i&7),  n = nt*16 + (lane&15)
// ---------------------------------------------------------------------------
__global__ void pack_B_kernel(const float* __restrict__ B, int ldb, int K, int Ncols,
                              _Float16* __restrict__ out, int ksteps, int ntiles_round) {
    int t = blockIdx.x * blockDim.x + threadIdx.x;
    int total = ntiles_round * ksteps * 32 * 16;
    if (t >= total) return;
    int i    = t & 15;
    int lane = (t >> 4) & 31;
    int ks   = (t >> 9) % ksteps;
    int nt   = (t >> 9) / ksteps;
    int k = ks * 32 + ((i & 8) ? 16 : 0) + ((lane >> 4) << 3) + (i & 7);
    int n = nt * 16 + (lane & 15);
    out[t] = (k < K && n < Ncols) ? (_Float16)B[(size_t)k * ldb + n] : (_Float16)0.0f;
}

// Zero-pad pi [G,625] -> pipad [G,640]
__global__ void pad_pi_kernel(const float* __restrict__ pi, float* __restrict__ pipad,
                              int rows, int K, int Kpad) {
    int t = blockIdx.x * blockDim.x + threadIdx.x;
    if (t >= rows * Kpad) return;
    int r = t / Kpad, k = t - r * Kpad;
    pipad[t] = (k < K) ? pi[(size_t)r * K + k] : 0.0f;
}

// ---------------------------------------------------------------------------
// WMMA GEMM: C[M,Ncols] = A[M,K] @ B + bias.  A is f32 row-major (M % 16 == 0,
// K % 32 == 0, lda % 4 == 0), B pre-packed f16.  One wave per 16-row strip x
// 4 column tiles: A loaded once per k-step (4 x float4), 4 v_wmma per step.
// ---------------------------------------------------------------------------
__global__ void wmma_gemm_f16(const float* __restrict__ A, int lda,
                              const _Float16* __restrict__ Bpk,
                              const float* __restrict__ bias,
                              float* __restrict__ C, int ldc,
                              int ksteps, int Ncols) {
    const int lane = threadIdx.x & 31;
    const int half = lane >> 4;
    const int l15  = lane & 15;
    const size_t arow = (size_t)(blockIdx.x * 16 + l15) * lda;
    const int nt0 = blockIdx.y * 4;

    v8f acc[4] = {{}, {}, {}, {}};
    for (int ks = 0; ks < ksteps; ++ks) {
        const float* ap = A + arow + ks * 32 + half * 8;
        float4 a0 = *(const float4*)(ap);
        float4 a1 = *(const float4*)(ap + 4);
        float4 a2 = *(const float4*)(ap + 16);
        float4 a3 = *(const float4*)(ap + 20);
        v16h a;
        a[0]  = (_Float16)a0.x; a[1]  = (_Float16)a0.y;
        a[2]  = (_Float16)a0.z; a[3]  = (_Float16)a0.w;
        a[4]  = (_Float16)a1.x; a[5]  = (_Float16)a1.y;
        a[6]  = (_Float16)a1.z; a[7]  = (_Float16)a1.w;
        a[8]  = (_Float16)a2.x; a[9]  = (_Float16)a2.y;
        a[10] = (_Float16)a2.z; a[11] = (_Float16)a2.w;
        a[12] = (_Float16)a3.x; a[13] = (_Float16)a3.y;
        a[14] = (_Float16)a3.z; a[15] = (_Float16)a3.w;
        const _Float16* bp = Bpk + ((size_t)(nt0 * ksteps + ks) * 32 + lane) * 16;
#pragma unroll
        for (int j = 0; j < 4; ++j) {
            v16h b = *(const v16h*)(bp + (size_t)j * ksteps * 512);
            acc[j] = __builtin_amdgcn_wmma_f32_16x16x32_f16(
                false, a, false, b, (short)0, acc[j], false, false);
        }
    }
#pragma unroll
    for (int j = 0; j < 4; ++j) {
        int nn = (nt0 + j) * 16 + l15;
        if (nn < Ncols) {
            float bv = bias ? bias[nn] : 0.0f;
#pragma unroll
            for (int r = 0; r < 8; ++r) {
                int mm = blockIdx.x * 16 + r + half * 8;
                C[(size_t)mm * ldc + nn] = acc[j][r] + bv;
            }
        }
    }
}

// ---------------------------------------------------------------------------
// agg[i,f] = hw[i,f] * snorm[i]   (self-loop contribution, also init)
// ---------------------------------------------------------------------------
__global__ void selfinit_kernel(const float* __restrict__ hw,
                                const float* __restrict__ snorm,
                                float* __restrict__ agg, int n) {
    int t = blockIdx.x * blockDim.x + threadIdx.x;
    if (t < n * 64) agg[t] = hw[t] * snorm[t >> 6];
}

// ---------------------------------------------------------------------------
// Edge scatter: agg[col,f] += hw[row,f] * dis[row]*dis[col]
// lane = feature -> coalesced loads + consecutive-address f32 atomics in L2.
// ---------------------------------------------------------------------------
__global__ void edge_scatter_kernel(const int* __restrict__ row, const int* __restrict__ col,
                                    const float* __restrict__ dis,
                                    const float* __restrict__ hw,
                                    float* __restrict__ agg, long long total) {
    long long t = (long long)blockIdx.x * blockDim.x + threadIdx.x;
    if (t >= total) return;
    int e = (int)(t >> 6);
    int f = (int)(t & 63);
    int r = row[e], c = col[e];
    float w = dis[r] * dis[c];
    atomicAdd(&agg[(size_t)c * 64 + f], hw[(size_t)r * 64 + f] * w);
}

// ---------------------------------------------------------------------------
// h = relu(agg + b); pool[batch[i], layer*64+f] = max(pool, h)  (jk+maxpool
// fused; post-ReLU values are >=0 so int atomicMax on float bits is exact)
// ---------------------------------------------------------------------------
__global__ void finalize_kernel(const float* __restrict__ agg, const float* __restrict__ bias,
                                const int* __restrict__ batch,
                                float* __restrict__ hbuf, float* __restrict__ pool,
                                int layer, int n) {
    int t = blockIdx.x * blockDim.x + threadIdx.x;
    if (t >= n * 64) return;
    int i = t >> 6, f = t & 63;
    float v = fmaxf(agg[t] + bias[f], 0.0f);
    hbuf[t] = v;
    atomicMax((int*)&pool[(size_t)batch[i] * 320 + layer * 64 + f], __float_as_int(v));
}

// ---------------------------------------------------------------------------
// Row-wise LayerNorm (+ optional ReLU).  One block per row.
// ---------------------------------------------------------------------------
__global__ void layernorm_kernel(const float* __restrict__ X, int ldx, int D,
                                 const float* __restrict__ gamma, const float* __restrict__ beta,
                                 float* __restrict__ Y, int ldy, int do_relu) {
    __shared__ float sh1[256];
    __shared__ float sh2[256];
    int row = blockIdx.x;
    const float* x = X + (size_t)row * ldx;
    float s = 0.f, s2 = 0.f;
    for (int j = threadIdx.x; j < D; j += blockDim.x) {
        float v = x[j]; s += v; s2 += v * v;
    }
    sh1[threadIdx.x] = s; sh2[threadIdx.x] = s2;
    __syncthreads();
    for (int o = blockDim.x >> 1; o > 0; o >>= 1) {
        if (threadIdx.x < o) {
            sh1[threadIdx.x] += sh1[threadIdx.x + o];
            sh2[threadIdx.x] += sh2[threadIdx.x + o];
        }
        __syncthreads();
    }
    float mean = sh1[0] / D;
    float var  = sh2[0] / D - mean * mean;
    float rstd = rsqrtf(var + 1e-5f);
    float* y = Y + (size_t)row * ldy;
    for (int j = threadIdx.x; j < D; j += blockDim.x) {
        float v = (x[j] - mean) * rstd * gamma[j] + beta[j];
        if (do_relu) v = fmaxf(v, 0.0f);
        y[j] = v;
    }
}

// ---------------------------------------------------------------------------
extern "C" void kernel_launch(void* const* d_in, const int* in_sizes, int n_in,
                              void* d_out, int out_size, void* d_ws, size_t ws_size,
                              hipStream_t stream) {
    (void)n_in; (void)out_size; (void)ws_size;

    const int N = in_sizes[2];          // 100000
    const int E = in_sizes[1] / 2;      // 1600000
    const int G = 64, H = 64, L = 5, P = 625, PPAD = 640;
    const int DG = 512, DP = 512, DOUT = 1000;
    const int LH = L * H;               // 320
    const int DEMB = DG + DP;           // 1024

    const float* x       = (const float*)d_in[0];
    const int*   ei      = (const int*)  d_in[1];
    const int*   batch   = (const int*)  d_in[2];
    const float* pi      = (const float*)d_in[3];
    const float* W0      = (const float*)d_in[4];
    const float* b0      = (const float*)d_in[5];
    const float* Ws      = (const float*)d_in[6];
    const float* bs      = (const float*)d_in[7];
    const float* Wg      = (const float*)d_in[8];
    const float* bg      = (const float*)d_in[9];
    const float* g_gamma = (const float*)d_in[10];
    const float* g_beta  = (const float*)d_in[11];
    const float* Wp      = (const float*)d_in[12];
    const float* bp      = (const float*)d_in[13];
    const float* p_gamma = (const float*)d_in[14];
    const float* p_beta  = (const float*)d_in[15];
    const float* Wf      = (const float*)d_in[16];
    const float* bf      = (const float*)d_in[17];
    const float* f_gamma = (const float*)d_in[18];
    const float* f_beta  = (const float*)d_in[19];
    float* out = (float*)d_out;

    const int* row = ei;       // messages flow row -> col
    const int* col = ei + E;

    // ---- workspace carve-up ----
    float* w = (float*)d_ws;
    float* degacc = w;                w += N;
    float* dis    = w;                w += N;
    float* snorm  = w;                w += N;
    float* hw     = w;                w += (size_t)N * H;
    float* agg    = w;                w += (size_t)N * H;
    float* hbuf   = w;                w += (size_t)N * H;
    float* pool   = w;                w += (size_t)G * LH;
    float* t1     = w;                w += (size_t)G * DG;
    float* t2     = w;                w += (size_t)G * DP;
    float* emb    = w;                w += (size_t)G * DEMB;
    float* t3     = w;                w += (size_t)G * DOUT;
    float* pipad  = w;                w += (size_t)G * PPAD;
    _Float16* hp  = (_Float16*)w;
    // packed weights (sizes: ntiles_round * ksteps * 512 halves)
    const int conv_ksteps = H / 32;                  // 2
    const int conv_pk     = 4 * conv_ksteps * 512;   // 4096 halves per layer
    _Float16* convpk = hp;            hp += (size_t)L * conv_pk;
    const int g_ksteps = LH / 32;                    // 10
    _Float16* wgpk   = hp;            hp += (size_t)(DG / 16) * g_ksteps * 512;
    const int p_ksteps = PPAD / 32;                  // 20
    _Float16* wppk   = hp;            hp += (size_t)(DP / 16) * p_ksteps * 512;
    const int f_ksteps = DEMB / 32;                  // 32
    const int f_ntr    = 64;                         // 63 tiles rounded to 4
    _Float16* wfpk   = hp;            hp += (size_t)f_ntr * f_ksteps * 512;

    // ---- pack all weights into WMMA order ----
    for (int l = 0; l < L; ++l) {
        const float* Wl = (l == 0) ? W0 : (Ws + (size_t)(l - 1) * H * H);
        pack_B_kernel<<<(conv_pk + 255) / 256, 256, 0, stream>>>(
            Wl, H, H, H, convpk + (size_t)l * conv_pk, conv_ksteps, 4);
    }
    {
        int tot = (DG / 16) * g_ksteps * 512;
        pack_B_kernel<<<(tot + 255) / 256, 256, 0, stream>>>(Wg, DG, LH, DG, wgpk, g_ksteps, DG / 16);
        tot = (DP / 16) * p_ksteps * 512;
        pack_B_kernel<<<(tot + 255) / 256, 256, 0, stream>>>(Wp, DP, P, DP, wppk, p_ksteps, DP / 16);
        tot = f_ntr * f_ksteps * 512;
        pack_B_kernel<<<(tot + 255) / 256, 256, 0, stream>>>(Wf, DOUT, DEMB, DOUT, wfpk, f_ksteps, f_ntr);
        pad_pi_kernel<<<(G * PPAD + 255) / 256, 256, 0, stream>>>(pi, pipad, G, P, PPAD);
    }

    // ---- degrees / normalization ----
    hipMemsetAsync(degacc, 0, (size_t)N * sizeof(float), stream);
    hipMemsetAsync(pool,   0, (size_t)G * LH * sizeof(float), stream);
    deg_kernel<<<(E + 255) / 256, 256, 0, stream>>>(col, degacc, E);
    norm_kernel<<<(N + 255) / 256, 256, 0, stream>>>(degacc, dis, snorm, N);

    // ---- 5 GCN layers, max-pool fused ----
    const long long etotal = (long long)E * H;
    const int eblocks = (int)((etotal + 255) / 256);
    const int nfeat_blocks = (N * H + 255) / 256;
    dim3 gconv(N / 16, 1);   // 4 column tiles per wave covers all 64 cols

    for (int l = 0; l < L; ++l) {
        const float* hin = (l == 0) ? x : hbuf;
        const float* bl  = (l == 0) ? b0 : (bs + (size_t)(l - 1) * H);

        wmma_gemm_f16<<<gconv, 32, 0, stream>>>(hin, H, convpk + (size_t)l * conv_pk,
                                                nullptr, hw, H, conv_ksteps, H);
        selfinit_kernel<<<nfeat_blocks, 256, 0, stream>>>(hw, snorm, agg, N);
        edge_scatter_kernel<<<eblocks, 256, 0, stream>>>(row, col, dis, hw, agg, etotal);
        finalize_kernel<<<nfeat_blocks, 256, 0, stream>>>(agg, bl, batch, hbuf, pool, l, N);
    }

    // ---- graph embedding head:  relu(LN(pool @ Wg + bg)) -> emb[:, :512] ----
    wmma_gemm_f16<<<dim3(G / 16, DG / 64), 32, 0, stream>>>(pool, LH, wgpk, bg, t1, DG, g_ksteps, DG);
    layernorm_kernel<<<G, 256, 0, stream>>>(t1, DG, DG, g_gamma, g_beta, emb, DEMB, 1);

    // ---- persistence head:  relu(LN(pi @ Wp + bp)) -> emb[:, 512:] ----
    wmma_gemm_f16<<<dim3(G / 16, DP / 64), 32, 0, stream>>>(pipad, PPAD, wppk, bp, t2, DP, p_ksteps, DP);
    layernorm_kernel<<<G, 256, 0, stream>>>(t2, DP, DP, p_gamma, p_beta, emb + DG, DEMB, 1);

    // ---- fc head:  LN(emb @ Wf + bf) -> out ----
    wmma_gemm_f16<<<dim3(G / 16, f_ntr / 4), 32, 0, stream>>>(emb, DEMB, wfpk, bf, t3, DOUT, f_ksteps, DOUT);
    layernorm_kernel<<<G, 256, 0, stream>>>(t3, DOUT, DOUT, f_gamma, f_beta, out, DOUT, 0);
}
